// GraphSAGE_50319836840200
// MI455X (gfx1250) — matrix operand order repaired and checked by reading, not compiled, not run on hardware
//
#include <hip/hip_runtime.h>

// ---------------------------------------------------------------------------
// GraphSAGE (2-layer, mean aggr) + link scoring for MI455X (gfx1250, wave32)
//   layer: out = relu?( (agg/deg) @ W_l + b + x @ W_r )
// Dense GEMMs: v_wmma_f32_16x16x32_bf16, weights pre-swizzled to fragment
// layout in bf16 (one-time pack), activations staged fp32->bf16 in LDS.
// Sparse aggregation: L2-resident global_atomic_add_f32.
// ---------------------------------------------------------------------------

typedef __attribute__((ext_vector_type(16))) __bf16        v16bf;
typedef __attribute__((ext_vector_type(8)))  float         v8f;
typedef __attribute__((ext_vector_type(8)))  unsigned int  v8u;
typedef __attribute__((ext_vector_type(4)))  unsigned int  v4u;
typedef __attribute__((ext_vector_type(2)))  unsigned int  v2u;

struct Frag32 { v4u lo, hi; };   // 32 bytes == one v16bf

#define DIN 128
#define DH  128

__device__ __forceinline__ unsigned short f2bf(float f) {
    // round-to-nearest-even f32 -> bf16
    unsigned int u = __builtin_bit_cast(unsigned int, f);
    u += 0x7FFFu + ((u >> 16) & 1u);
    return (unsigned short)(u >> 16);
}
__device__ __forceinline__ unsigned int pack2bf(float a, float b) {
    return (unsigned int)f2bf(a) | ((unsigned int)f2bf(b) << 16);
}

// ---------------------------------------------------------------------------
__global__ __launch_bounds__(256) void zero_kernel(float* __restrict__ p, long long n) {
    long long i = (long long)blockIdx.x * blockDim.x + threadIdx.x;
    long long stride = (long long)gridDim.x * blockDim.x;
    for (; i < n; i += stride) p[i] = 0.0f;
}

__global__ __launch_bounds__(256) void degree_kernel(const int* __restrict__ dst,
                                                     float* __restrict__ deg, int E) {
    int e = blockIdx.x * blockDim.x + threadIdx.x;
    if (e < E) atomicAdd(&deg[dst[e]], 1.0f);
}

// One edge handled by 32 threads, 4 floats each (D == 128).
__global__ __launch_bounds__(256) void scatter_add_kernel(const float* __restrict__ feat,
                                                          const int* __restrict__ src,
                                                          const int* __restrict__ dst,
                                                          float* __restrict__ agg, int E) {
    long long t = (long long)blockIdx.x * blockDim.x + threadIdx.x;
    long long e = t >> 5;
    if (e >= E) return;
    int c4 = ((int)t & 31) * 4;
    int s = src[e], d = dst[e];
    const float4 v = *(const float4*)(feat + (long long)s * 128 + c4);
    float* ap = agg + (long long)d * 128 + c4;
    atomicAdd(ap + 0, v.x);
    atomicAdd(ap + 1, v.y);
    atomicAdd(ap + 2, v.z);
    atomicAdd(ap + 3, v.w);
}

// ---------------------------------------------------------------------------
// Pack W [128 x OUT] f32 row-major into per-lane WMMA B-fragments (bf16):
//   fragment (tile, kb) for lane l: N = tile*16 + (l&15), VGPR v holds
//   K rows kb*32 + (l>>4)*16 + 2v, +1.  8 dwords per lane, 32B aligned.
__global__ __launch_bounds__(256) void pack_weight_kernel(const float* __restrict__ W,
                                                          unsigned int* __restrict__ out,
                                                          int OUT) {
    int t = blockIdx.x * blockDim.x + threadIdx.x;
    int total = (OUT / 16) * 4 * 32;            // tiles * kb * lanes
    if (t >= total) return;
    int lane = t & 31, kb = (t >> 5) & 3, tile = t >> 7;
    int g = lane >> 4, nl = lane & 15;
    int col = tile * 16 + nl;
    unsigned int* dst = out + (long long)t * 8;
#pragma unroll
    for (int v = 0; v < 8; ++v) {
        int k0 = kb * 32 + g * 16 + 2 * v;
        dst[v] = pack2bf(W[(long long)k0 * OUT + col],
                         W[(long long)(k0 + 1) * OUT + col]);
    }
}

// ---------------------------------------------------------------------------
// Fused mean + dual-GEMM + bias (+relu). Block covers 32 nodes (two M-tiles);
// blockDim.x = (OUT/16)*32, each wave owns one 16-col N-tile. K fixed at 128.
// pWl/pWr are pre-packed bf16 fragment buffers; bl is the f32 bias.
__global__ __launch_bounds__(256) void sage_wmma_layer(const float* __restrict__ xin,
                                                       const float* __restrict__ agg,
                                                       const float* __restrict__ deg,
                                                       const unsigned int* __restrict__ pWl,
                                                       const float* __restrict__ bl,
                                                       const unsigned int* __restrict__ pWr,
                                                       float* __restrict__ out,
                                                       int n_nodes, int OUT, int do_relu) {
    __shared__ unsigned int sMean[32 * 64];   // 32 rows x 128 bf16 (as dwords)
    __shared__ unsigned int sX[32 * 64];
    __shared__ float sInv[32];

    const int tid = threadIdx.x;
    const int nb  = blockIdx.x * 32;

    if (tid < 32) {
        int node = nb + tid;
        float d = (node < n_nodes) ? deg[node] : 1.0f;
        sInv[tid] = 1.0f / (d > 1.0f ? d : 1.0f);
    }
    __syncthreads();

    // Stage 32x128 mean/x tiles as bf16 (vectorized: float4 in, 2 dwords out).
    for (int i = tid; i < 32 * 32; i += blockDim.x) {   // 32 rows x 32 col-quads
        int r = i >> 5, c4 = (i & 31) * 4;
        int node = nb + r;
        float4 am = make_float4(0.f, 0.f, 0.f, 0.f);
        float4 av = am;
        if (node < n_nodes) {
            long long off = (long long)node * 128 + c4;
            am = *(const float4*)(agg + off);
            float inv = sInv[r];
            am.x *= inv; am.y *= inv; am.z *= inv; am.w *= inv;
            av = *(const float4*)(xin + off);
        }
        int di = r * 64 + (c4 >> 1);
        v2u pm = { pack2bf(am.x, am.y), pack2bf(am.z, am.w) };
        v2u pv = { pack2bf(av.x, av.y), pack2bf(av.z, av.w) };
        *(v2u*)(sMean + di) = pm;
        *(v2u*)(sX + di)    = pv;
    }
    __syncthreads();

    const int wave = tid >> 5;
    const int lane = tid & 31;
    const int g    = lane >> 4;
    const int nl   = lane & 15;

    v8f c0 = {}, c1 = {};
#pragma unroll
    for (int kb = 0; kb < 4; ++kb) {
        // B fragments: single 32B load each from the pre-packed buffers.
        long long fb = ((long long)(wave * 4 + kb) * 32 + lane) * 8;
        v16bf fbl = __builtin_bit_cast(v16bf, *(const v8u*)(pWl + fb));
        v16bf fbr = __builtin_bit_cast(v16bf, *(const v8u*)(pWr + fb));

        // A fragments: two 16B LDS loads each, bit_cast straight to v16bf.
        // Row m, dword base m*64 + kb*16 + g*8; halves at +0 and +4 dwords.
        int ab0 = nl * 64 + kb * 16 + g * 8;          // M-tile 0: rows 0..15
        int ab1 = (nl + 16) * 64 + kb * 16 + g * 8;   // M-tile 1: rows 16..31
        Frag32 fm0 = { *(const v4u*)(sMean + ab0), *(const v4u*)(sMean + ab0 + 4) };
        Frag32 fm1 = { *(const v4u*)(sMean + ab1), *(const v4u*)(sMean + ab1 + 4) };
        Frag32 fx0 = { *(const v4u*)(sX + ab0),    *(const v4u*)(sX + ab0 + 4) };
        Frag32 fx1 = { *(const v4u*)(sX + ab1),    *(const v4u*)(sX + ab1 + 4) };

        c0 = __builtin_amdgcn_wmma_f32_16x16x32_bf16(false, __builtin_bit_cast(v16bf, fm0),
                                                     false, fbl, (short)0, c0, false, false);
        c1 = __builtin_amdgcn_wmma_f32_16x16x32_bf16(false, __builtin_bit_cast(v16bf, fm1),
                                                     false, fbl, (short)0, c1, false, false);
        c0 = __builtin_amdgcn_wmma_f32_16x16x32_bf16(false, __builtin_bit_cast(v16bf, fx0),
                                                     false, fbr, (short)0, c0, false, false);
        c1 = __builtin_amdgcn_wmma_f32_16x16x32_bf16(false, __builtin_bit_cast(v16bf, fx1),
                                                     false, fbr, (short)0, c1, false, false);
    }

    // Epilogue: C/D layout — VGPR r: lanes 0-15 -> M=r, lanes 16-31 -> M=r+8.
    const int col = wave * 16 + nl;
    const float bias = bl[col];
#pragma unroll
    for (int r = 0; r < 8; ++r) {
        int m = g ? (r + 8) : r;
        float v0 = c0[r] + bias;
        float v1 = c1[r] + bias;
        if (do_relu) {
            v0 = v0 > 0.0f ? v0 : 0.0f;
            v1 = v1 > 0.0f ? v1 : 0.0f;
        }
        int n0 = nb + m, n1 = nb + 16 + m;
        if (n0 < n_nodes) out[(long long)n0 * OUT + col] = v0;
        if (n1 < n_nodes) out[(long long)n1 * OUT + col] = v1;
    }
}

// ---------------------------------------------------------------------------
// pos/neg scoring: out[t] = dot(h[a], h[b]) over 64 dims; t<EP pos, else neg.
__global__ __launch_bounds__(256) void score_kernel(const float* __restrict__ h,
                                                    const int* __restrict__ pos,
                                                    const int* __restrict__ neg,
                                                    float* __restrict__ out, int EP) {
    int t = blockIdx.x * blockDim.x + threadIdx.x;
    if (t >= 2 * EP) return;
    const int* ei = (t < EP) ? pos : neg;
    int e = (t < EP) ? t : (t - EP);
    int a = ei[e], b = ei[EP + e];
    const float4* pa = (const float4*)(h + (long long)a * 64);
    const float4* pb = (const float4*)(h + (long long)b * 64);
    float s = 0.0f;
#pragma unroll
    for (int i = 0; i < 16; ++i) {
        float4 va = pa[i], vb = pb[i];
        s += va.x * vb.x + va.y * vb.y + va.z * vb.z + va.w * vb.w;
    }
    out[t] = s;
}

// ---------------------------------------------------------------------------
extern "C" void kernel_launch(void* const* d_in, const int* in_sizes, int n_in,
                              void* d_out, int out_size, void* d_ws, size_t ws_size,
                              hipStream_t stream) {
    const float* x    = (const float*)d_in[0];
    const int*   edge = (const int*)d_in[1];
    const int*   pos  = (const int*)d_in[2];
    const int*   neg  = (const int*)d_in[3];
    const float* Wl0  = (const float*)d_in[4];
    const float* bl0  = (const float*)d_in[5];
    const float* Wr0  = (const float*)d_in[6];
    const float* Wl1  = (const float*)d_in[7];
    const float* bl1  = (const float*)d_in[8];
    const float* Wr1  = (const float*)d_in[9];

    const int N  = in_sizes[0] / DIN;   // 100000
    const int E  = in_sizes[1] / 2;     // 1600000
    const int EP = in_sizes[2] / 2;     // 500000

    // Workspace (floats): agg[N*128] | deg[N] | h0[N*128] | h1[N*64] | packed W
    float* agg = (float*)d_ws;
    float* deg = agg + (long long)N * 128;
    float* h0  = deg + N;
    float* h1  = h0 + (long long)N * 128;
    unsigned int* pWl0 = (unsigned int*)(h1 + (long long)N * 64);
    unsigned int* pWr0 = pWl0 + 128 * 128 / 2;   // bf16 pairs -> dwords
    unsigned int* pWl1 = pWr0 + 128 * 128 / 2;
    unsigned int* pWr1 = pWl1 + 128 * 64 / 2;

    const int* src = edge;
    const int* dst = edge + E;

    const long long scatterThreads = (long long)E * 32;
    const int scatterBlocks = (int)((scatterThreads + 255) / 256);
    const int nodeTiles = (N + 31) / 32;

    // --- One-time weight packing (cheap; runs every call for determinism) ---
    pack_weight_kernel<<<4, 256, 0, stream>>>(Wl0, pWl0, 128);  // 1024 frag-lanes
    pack_weight_kernel<<<4, 256, 0, stream>>>(Wr0, pWr0, 128);
    pack_weight_kernel<<<2, 256, 0, stream>>>(Wl1, pWl1, 64);   // 512 frag-lanes
    pack_weight_kernel<<<2, 256, 0, stream>>>(Wr1, pWr1, 64);

    // --- Layer 0 ---
    zero_kernel<<<2048, 256, 0, stream>>>(agg, (long long)N * 128 + N);
    degree_kernel<<<(E + 255) / 256, 256, 0, stream>>>(dst, deg, E);
    scatter_add_kernel<<<scatterBlocks, 256, 0, stream>>>(x, src, dst, agg, E);
    sage_wmma_layer<<<nodeTiles, 256, 0, stream>>>(x, agg, deg, pWl0, bl0, pWr0,
                                                   h0, N, 128, 1);

    // --- Layer 1 ---
    zero_kernel<<<2048, 256, 0, stream>>>(agg, (long long)N * 128);
    scatter_add_kernel<<<scatterBlocks, 256, 0, stream>>>(h0, src, dst, agg, E);
    sage_wmma_layer<<<nodeTiles, 128, 0, stream>>>(h0, agg, deg, pWl1, bl1, pWr1,
                                                   h1, N, 64, 0);

    // --- Scoring ---
    score_kernel<<<(2 * EP + 255) / 256, 256, 0, stream>>>(h1, pos, neg,
                                                           (float*)d_out, EP);
}
